// PrefillQAttention_27247272526200
// MI455X (gfx1250) — matrix operand order repaired and checked by reading, not compiled, not run on hardware
//
#include <hip/hip_runtime.h>
#include <hip/hip_bf16.h>

typedef __attribute__((ext_vector_type(16))) _Float16 v16h;
typedef __attribute__((ext_vector_type(8)))  float    v8f;

#define BATCH 2
#define HEADS 16
#define SEQ   2048
#define DH    128
#define NBH   (BATCH*HEADS)          // 32 heads
#define ROWS  (NBH*SEQ)              // 65536 rows per tensor
#define NPROJ 7
#define NBLK  (ROWS/32)              // 2048 blocks of 32
#define NKB   (SEQ/32)               // 64 key blocks per head
#define VTP   40                     // padded Vt row (halves): 80B rows, 16B aligned
#define SCALE 0.08838834764831845f   // 1/sqrt(128)

// ---------------------------------------------------------------------------
// 1) LSH hash: one wave per row; Gray permutation == code ^ (code>>1).
// ---------------------------------------------------------------------------
__global__ __launch_bounds__(256) void hash_kernel(const float* __restrict__ q,
                                                   const float* __restrict__ k,
                                                   const float* __restrict__ proj,
                                                   int* __restrict__ qh,
                                                   int* __restrict__ kh) {
  int wid  = (blockIdx.x * blockDim.x + threadIdx.x) >> 5;   // 2*ROWS waves
  int lane = threadIdx.x & 31;
  int row  = wid & (ROWS - 1);
  const float* x    = (wid < ROWS) ? q  : k;
  int*         outp = (wid < ROWS) ? qh : kh;
  const float* xr = x + (size_t)row * DH;
  float acc[NPROJ];
#pragma unroll
  for (int p = 0; p < NPROJ; ++p) acc[p] = 0.f;
#pragma unroll
  for (int j = 0; j < 4; ++j) {
    int d = lane + 32 * j;
    float xv = xr[d];
#pragma unroll
    for (int p = 0; p < NPROJ; ++p) acc[p] += xv * proj[d * NPROJ + p];
  }
#pragma unroll
  for (int p = 0; p < NPROJ; ++p) {
    float a = acc[p];
    a += __shfl_xor(a, 1);  a += __shfl_xor(a, 2);  a += __shfl_xor(a, 4);
    a += __shfl_xor(a, 8);  a += __shfl_xor(a, 16);
    acc[p] = a;
  }
  if (lane == 0) {
    int code = 0;
#pragma unroll
    for (int p = 0; p < NPROJ; ++p) code |= (acc[p] > 0.f) ? (1 << p) : 0;
    outp[row] = code ^ (code >> 1);
  }
}

// ---------------------------------------------------------------------------
// 2) Stable counting sort of q_hash per head (values in [0,128)).
// ---------------------------------------------------------------------------
__global__ __launch_bounds__(256) void sort_kernel(const int* __restrict__ qh,
                                                   int* __restrict__ sidx,
                                                   int* __restrict__ qhs) {
  __shared__ int hsh[SEQ];
  __shared__ int cnt[128];
  __shared__ int start[128];
  int bh = blockIdx.x;
  int t  = threadIdx.x;
  if (t < 128) cnt[t] = 0;
  __syncthreads();
  for (int i = t; i < SEQ; i += 256) {
    int h = qh[bh * SEQ + i];
    hsh[i] = h;
    atomicAdd(&cnt[h], 1);
  }
  __syncthreads();
  if (t == 0) {
    int s = 0;
    for (int i = 0; i < 128; ++i) { start[i] = s; s += cnt[i]; }
  }
  __syncthreads();
  for (int i = t; i < SEQ; i += 256) {
    int h = hsh[i];
    int rank = start[h];
    for (int j = 0; j < i; ++j) rank += (hsh[j] == h) ? 1 : 0;
    sidx[bh * SEQ + rank] = i;
    qhs [bh * SEQ + rank] = h;
  }
}

// ---------------------------------------------------------------------------
// 3) Criticality per 32-block: keep = (match count <= 0)
// ---------------------------------------------------------------------------
__global__ __launch_bounds__(256) void crit_kernel(const int* __restrict__ kh,
                                                   const int* __restrict__ qhs,
                                                   float* __restrict__ keep) {
  int b = blockIdx.x * blockDim.x + threadIdx.x;
  if (b >= NBLK) return;
  int c = 0;
#pragma unroll
  for (int j = 0; j < 32; ++j) c += (kh[b * 32 + j] == qhs[b * 32 + j]) ? 1 : 0;
  keep[b] = (c <= 0) ? 1.f : 0.f;
}

// ---------------------------------------------------------------------------
// 4) Dense flash attention: v_wmma_f32_16x16x32_f16, double-buffered
//    GLOBAL_LOAD_ASYNC_TO_LDS_B128 staging (f32), single cooperative
//    f32->f16 convert pass per tile (Kh row-major, Vt transposed+padded)
//    so every WMMA B-fragment is a contiguous 16-half ds_load_b128 feed.
// ---------------------------------------------------------------------------
__device__ __forceinline__ void async_tile_b128(unsigned ldsOff,
                                                const float* gsrc,
                                                int tid) {
  // 32x128 f32 tile = 16KB = 1024 x b128; 8 async instrs per thread (8/wave).
#pragma unroll
  for (int i = 0; i < 8; ++i) {
    unsigned u = (unsigned)tid + 128u * i;
    unsigned lds = ldsOff + u * 16u;
    unsigned long long ga = (unsigned long long)(uintptr_t)gsrc + (unsigned long long)u * 16u;
    asm volatile("global_load_async_to_lds_b128 %0, %1, off"
                 :: "v"(lds), "v"(ga) : "memory");
  }
}

__global__ __launch_bounds__(128) void dense_flash(const float* __restrict__ q,
                                                   const float* __restrict__ k,
                                                   const float* __restrict__ v,
                                                   float* __restrict__ out) {
  extern __shared__ __align__(16) char smem[];
  float*    Kst = (float*)smem;                    // [2][32*DH] f32 async staging
  float*    Vst = Kst + 2 * 32 * DH;               // [2][32*DH] f32 async staging
  _Float16* Kh  = (_Float16*)(Vst + 2 * 32 * DH);  // [32][DH]   f16 row-major
  _Float16* Vt  = Kh + 32 * DH;                    // [DH][VTP]  f16 transposed
  __shared__ _Float16 Psh[4][16][32];              // per-wave P transpose staging

  const int tid   = threadIdx.x;
  const int lane  = tid & 31;
  const int wave  = tid >> 5;
  const int group = lane >> 4;
  const int l16   = lane & 15;
  const int bh    = blockIdx.y;
  const int q0    = blockIdx.x * 64;
  const size_t ho = (size_t)bh * SEQ * DH;

  const unsigned ldsK = (unsigned)(uintptr_t)(void*)Kst;   // DS byte address
  const unsigned ldsV = (unsigned)(uintptr_t)(void*)Vst;

  // --- Q fragments in A layout (scale folded into Q) ---
  v16h qA[4];
  {
    int qrow = q0 + wave * 16 + l16;
    const float* qp = q + ho + (size_t)qrow * DH;
#pragma unroll
    for (int c = 0; c < 4; ++c)
#pragma unroll
      for (int e = 0; e < 16; ++e) {
        int kk = 16 * group + e;   // A-matrix k striping
        qA[c][e] = (_Float16)(qp[32 * c + kk] * SCALE);
      }
  }

  v8f o[8];
#pragma unroll
  for (int i = 0; i < 8; ++i) o[i] = (v8f)(0.f);
  float mrow[8], lrow[8];
#pragma unroll
  for (int r = 0; r < 8; ++r) { mrow[r] = -1e30f; lrow[r] = 0.f; }

  // Prologue: async-stage tile 0 into buffer 0.
  async_tile_b128(ldsK, k + ho, tid);
  async_tile_b128(ldsV, v + ho, tid);

  for (int kb = 0; kb < NKB; ++kb) {
    const int cur = kb & 1;
    if (kb + 1 < NKB) {
      const int nxt = cur ^ 1;
      async_tile_b128(ldsK + (unsigned)nxt * 32u * DH * 4u, k + ho + (size_t)(kb + 1) * 32 * DH, tid);
      async_tile_b128(ldsV + (unsigned)nxt * 32u * DH * 4u, v + ho + (size_t)(kb + 1) * 32 * DH, tid);
      asm volatile("s_wait_asynccnt 0x10" ::: "memory");   // current tile resident
    } else {
      asm volatile("s_wait_asynccnt 0x0" ::: "memory");
    }
    __syncthreads();

    // --- one cooperative convert pass: f32 staging -> Kh (row) + Vt (transposed)
    {
      const float4* Ks4 = (const float4*)(Kst + (size_t)cur * 32 * DH);
      const float4* Vs4 = (const float4*)(Vst + (size_t)cur * 32 * DH);
      for (int i = tid; i < 32 * DH / 4; i += 128) {
        float4 kv = Ks4[i];
        float4 vv = Vs4[i];
        int r  = i >> 5;             // key row 0..31
        int c4 = (i & 31) * 4;       // d base
        Kh[r * DH + c4 + 0] = (_Float16)kv.x;
        Kh[r * DH + c4 + 1] = (_Float16)kv.y;
        Kh[r * DH + c4 + 2] = (_Float16)kv.z;
        Kh[r * DH + c4 + 3] = (_Float16)kv.w;
        Vt[(c4 + 0) * VTP + r] = (_Float16)vv.x;
        Vt[(c4 + 1) * VTP + r] = (_Float16)vv.y;
        Vt[(c4 + 2) * VTP + r] = (_Float16)vv.z;
        Vt[(c4 + 3) * VTP + r] = (_Float16)vv.w;
      }
    }
    __syncthreads();

    // --- scores S = (Q*scale) K^T : two 16-key tiles, 4 d-chunks of K=32
    v8f s0 = (v8f)(0.f), s1 = (v8f)(0.f);
#pragma unroll
    for (int c = 0; c < 4; ++c) {
      v16h b0, b1;
#pragma unroll
      for (int e = 0; e < 16; ++e) {
        int kk = 32 * c + 16 * group + e;           // B-matrix k striping
        b0[e] = Kh[l16 * DH + kk];
        b1[e] = Kh[(16 + l16) * DH + kk];
      }
      s0 = __builtin_amdgcn_wmma_f32_16x16x32_f16(false, qA[c], false, b0, (short)0, s0, false, false);
      s1 = __builtin_amdgcn_wmma_f32_16x16x32_f16(false, qA[c], false, b1, (short)0, s1, false, false);
    }

    // --- online softmax (rows on 16-lane halves; xor masks <= 8 stay in-half)
    float alpha[8];
#pragma unroll
    for (int r = 0; r < 8; ++r) {
      float mx = fmaxf(s0[r], s1[r]);
      mx = fmaxf(mx, __shfl_xor(mx, 1));
      mx = fmaxf(mx, __shfl_xor(mx, 2));
      mx = fmaxf(mx, __shfl_xor(mx, 4));
      mx = fmaxf(mx, __shfl_xor(mx, 8));
      float mnew = fmaxf(mrow[r], mx);
      float a  = __expf(mrow[r] - mnew);
      float p0 = __expf(s0[r] - mnew);
      float p1 = __expf(s1[r] - mnew);
      float rs = p0 + p1;
      rs += __shfl_xor(rs, 1);
      rs += __shfl_xor(rs, 2);
      rs += __shfl_xor(rs, 4);
      rs += __shfl_xor(rs, 8);
      lrow[r] = lrow[r] * a + rs;
      mrow[r] = mnew;
      alpha[r] = a;
      Psh[wave][r + 8 * group][l16]      = (_Float16)p0;
      Psh[wave][r + 8 * group][16 + l16] = (_Float16)p1;
    }
#pragma unroll
    for (int db = 0; db < 8; ++db)
#pragma unroll
      for (int r = 0; r < 8; ++r) o[db][r] *= alpha[r];

    // --- P back in A layout (per-wave staging; same-wave DS ordering)
    v16h pA;
#pragma unroll
    for (int e = 0; e < 16; ++e) {
      int kk = 16 * group + e;
      pA[e] = Psh[wave][l16][kk];
    }

    // --- O += P V  (Vt transposed: contiguous 16-half B fragments)
#pragma unroll
    for (int db = 0; db < 8; ++db) {
      v16h vB;
#pragma unroll
      for (int e = 0; e < 16; ++e) {
        vB[e] = Vt[(16 * db + l16) * VTP + 16 * group + e];
      }
      o[db] = __builtin_amdgcn_wmma_f32_16x16x32_f16(false, pA, false, vB, (short)0, o[db], false, false);
    }
    __syncthreads();   // done reading before next iter's convert/async overwrite
  }

  // --- normalize and write out (f32)
#pragma unroll
  for (int db = 0; db < 8; ++db)
#pragma unroll
    for (int r = 0; r < 8; ++r) {
      int row = q0 + wave * 16 + r + 8 * group;
      out[ho + (size_t)row * DH + 16 * db + l16] = o[db][r] / lrow[r];
    }
}

// ---------------------------------------------------------------------------
// 5) Blocked 32x32 critical attention + fused SSE partial vs out_actual.
// ---------------------------------------------------------------------------
__global__ __launch_bounds__(128) void blocked_kernel(const float* __restrict__ q,
                                                      const float* __restrict__ k,
                                                      const float* __restrict__ v,
                                                      const int* __restrict__ sidx,
                                                      const float* __restrict__ keep,
                                                      const float* __restrict__ outa,
                                                      float* __restrict__ partial) {
  __shared__ float Qs[32][DH];
  __shared__ float Ks[32][DH];
  __shared__ float Vs[32][DH];
  __shared__ float Ps[32][32];
  __shared__ float red[128];
  int nb = blockIdx.x;
  int bh = nb >> 6;
  int s0 = (nb & 63) * 32;
  int t  = threadIdx.x;
  float kf = keep[nb];
  const size_t ho = (size_t)bh * SEQ * DH;

  for (int idx = t; idx < 32 * DH; idx += 128) {
    int r = idx >> 7, c = idx & 127;
    int qsrc = sidx[bh * SEQ + s0 + r];
    Qs[r][c] = q[ho + (size_t)qsrc * DH + c];
    Ks[r][c] = k[ho + (size_t)(s0 + r) * DH + c] * kf;
    Vs[r][c] = v[ho + (size_t)(s0 + r) * DH + c];
  }
  __syncthreads();

  for (int p = t; p < 1024; p += 128) {
    int qi = p >> 5, ki = p & 31;
    float acc = 0.f;
    for (int d = 0; d < DH; ++d) acc += Qs[qi][d] * Ks[ki][d];
    Ps[qi][ki] = acc * SCALE;
  }
  __syncthreads();

  if (t < 32) {
    float mx = -1e30f;
    for (int j = 0; j < 32; ++j) mx = fmaxf(mx, Ps[t][j]);
    float s = 0.f;
    for (int j = 0; j < 32; ++j) { float e = __expf(Ps[t][j] - mx); Ps[t][j] = e; s += e; }
    float inv = 1.f / s;
    for (int j = 0; j < 32; ++j) Ps[t][j] *= inv;
  }
  __syncthreads();

  float sse = 0.f;
  for (int idx = t; idx < 32 * DH; idx += 128) {
    int qi = idx >> 7, d = idx & 127;
    float acc = 0.f;
    for (int j = 0; j < 32; ++j) acc += Ps[qi][j] * Vs[j][d];
    float diff = acc - outa[ho + (size_t)(s0 + qi) * DH + d];
    sse += diff * diff;
  }
  red[t] = sse;
  __syncthreads();
  for (int off = 64; off > 0; off >>= 1) {
    if (t < off) red[t] += red[t + off];
    __syncthreads();
  }
  if (t == 0) partial[nb] = red[0];
}

// ---------------------------------------------------------------------------
// 6) Final deterministic loss reduction (fixed summation order, no atomics).
// ---------------------------------------------------------------------------
__global__ __launch_bounds__(256) void loss_kernel(const float* __restrict__ partial,
                                                   float* __restrict__ outLoss) {
  __shared__ float red[256];
  int t = threadIdx.x;
  float s = 0.f;
  for (int i = t; i < NBLK; i += 256) s += partial[i];
  red[t] = s;
  __syncthreads();
  for (int off = 128; off > 0; off >>= 1) {
    if (t < off) red[t] += red[t + off];
    __syncthreads();
  }
  if (t == 0) outLoss[0] = red[0] / (float)((size_t)ROWS * DH);
}

// ---------------------------------------------------------------------------
extern "C" void kernel_launch(void* const* d_in, const int* in_sizes, int n_in,
                              void* d_out, int out_size, void* d_ws, size_t ws_size,
                              hipStream_t stream) {
  const float* q    = (const float*)d_in[0];
  const float* k    = (const float*)d_in[1];
  const float* v    = (const float*)d_in[2];
  const float* proj = (const float*)d_in[3];
  // d_in[4] = mask: all-true broadcast scalar, ignored.
  float* out = (float*)d_out;   // [B,H,S,D] flat, then 1 loss element

  char* ws = (char*)d_ws;
  int*   qh      = (int*)(ws);
  int*   kh      = (int*)(ws + (size_t)ROWS * 4);
  int*   sidx    = (int*)(ws + (size_t)ROWS * 8);
  int*   qhs     = (int*)(ws + (size_t)ROWS * 12);
  float* keep    = (float*)(ws + (size_t)ROWS * 16);
  float* partial = (float*)(ws + (size_t)ROWS * 16 + (size_t)NBLK * 4);

  // dynamic LDS for dense_flash: 2x16KB K + 2x16KB V staging + 8KB Kh + 10KB Vt
  const size_t dynLds = (size_t)(2 * 32 * DH + 2 * 32 * DH) * 4 + (size_t)32 * DH * 2 + (size_t)DH * VTP * 2;

  hash_kernel<<<(2 * ROWS) / 8, 256, 0, stream>>>(q, k, proj, qh, kh);   // 1 wave per row
  sort_kernel<<<NBH, 256, 0, stream>>>(qh, sidx, qhs);
  crit_kernel<<<(NBLK + 255) / 256, 256, 0, stream>>>(kh, qhs, keep);
  dense_flash<<<dim3(SEQ / 64, NBH), 128, dynLds, stream>>>(q, k, v, out);
  blocked_kernel<<<NBLK, 128, 0, stream>>>(q, k, v, sidx, keep, out, partial);
  loss_kernel<<<1, 256, 0, stream>>>(partial, out + (size_t)ROWS * DH);
}
